// SSM8Block_43800076484892
// MI455X (gfx1250) — compile-verified, matrix operand order checked by reading, not run on hardware
//
#include <hip/hip_runtime.h>
#include <hip/hip_bf16.h>

// ---------------------------------------------------------------------------
// Mamba SSM block on gfx1250 (MI455X), wave32 + WMMA.
// GEMMs: bf16 A/B, f32 accumulate via v_wmma_f32_16x16x32_bf16,
//        register-blocked 64x32 per wave (8 accumulators) for A/B-frag reuse.
// Scan:  2-pass chunked parallel scan (diagonal dA => chunk decay = exp(a*sum_dt)).
// ---------------------------------------------------------------------------

#define D_MODEL 640
#define D_INNER 1280
#define D_STATE 34
#define D_CONV  4
#define DT_RANK 40
#define BATCH   2
#define SEQ     2048
#define MTOT    (BATCH * SEQ)      // 4096
#define XDCOLS  (DT_RANK + 2 * D_STATE)  // 108
#define KPAD    64                 // dt GEMM K padded 40 -> 64
#define CHUNK   64
#define NCHUNK  (SEQ / CHUNK)      // 32

typedef __bf16  bf16_t;
typedef __attribute__((ext_vector_type(16))) __bf16 v16bf;
typedef __attribute__((ext_vector_type(8)))  __bf16 v8bf;
typedef __attribute__((ext_vector_type(8)))  float  v8f;

// ---------------------------------------------------------------------------
// Tiled WMMA GEMM: C[M,N] = A[M,K] * Bw[N,K]^T  (both row-major, bf16).
// Block: 256 thr = 8 waves arranged 2(M) x 4(N) -> 128x128 block tile.
// Wave: 64(M) x 32(N) = 4x2 sub-tiles of 16x16, 8 f32x8 accumulators.
// Per K-step (K+=32): 4 A-frags + 2 B-frags (12 b128 loads) feed 8 WMMAs.
// Frag layout (ISA 7.12.2, 16-bit A 16x32): lane%16 = M row (A) / N col (B),
// lane/16 selects K-half; element[0..7]=K(k0+h*8..), element[8..15]=K(+16).
// C/D: lane%16 = N col, row = 8*(lane/16) + vgpr.
// ---------------------------------------------------------------------------
__global__ __launch_bounds__(256)
void gemm_bf16_wmma(const bf16_t* __restrict__ A, const bf16_t* __restrict__ Bw,
                    float* __restrict__ C, int M, int N, int K) {
    const int lane = threadIdx.x & 31;
    const int wv   = threadIdx.x >> 5;
    const int sub  = lane & 15;   // M row (A) / N col (B, C)
    const int half = lane >> 4;   // K-half selector
    const int mbase = blockIdx.y * 128 + (wv & 1) * 64;
    const int nbase = blockIdx.x * 128 + (wv >> 1) * 32;

    const bf16_t* ap[4];
    #pragma unroll
    for (int i = 0; i < 4; ++i)
        ap[i] = A + (size_t)(mbase + i * 16 + sub) * K + half * 8;

    const bf16_t* bp[2];
    bool nval[2];
    #pragma unroll
    for (int j = 0; j < 2; ++j) {
        int ncol = nbase + j * 16 + sub;
        nval[j] = (ncol < N);
        if (ncol >= N) ncol = N - 1;     // clamp OOB loads (stores guarded)
        bp[j] = Bw + (size_t)ncol * K + half * 8;
    }

    v8f acc[4][2] = {};
    for (int k0 = 0; k0 < K; k0 += 32) {
        v16bf bfr[2];
        #pragma unroll
        for (int j = 0; j < 2; ++j) {
            v8bf blo = *(const v8bf*)(bp[j] + k0);
            v8bf bhi = *(const v8bf*)(bp[j] + k0 + 16);
            bfr[j] = __builtin_shufflevector(blo, bhi, 0,1,2,3,4,5,6,7,8,9,10,11,12,13,14,15);
        }
        #pragma unroll
        for (int i = 0; i < 4; ++i) {
            v8bf alo = *(const v8bf*)(ap[i] + k0);
            v8bf ahi = *(const v8bf*)(ap[i] + k0 + 16);
            v16bf af = __builtin_shufflevector(alo, ahi, 0,1,2,3,4,5,6,7,8,9,10,11,12,13,14,15);
            #pragma unroll
            for (int j = 0; j < 2; ++j) {
                acc[i][j] = __builtin_amdgcn_wmma_f32_16x16x32_bf16(
                    /*neg_a=*/false, af, /*neg_b=*/false, bfr[j],
                    /*c_mod=*/(short)0, acc[i][j],
                    /*reuse_a=*/false, /*reuse_b=*/false);
            }
        }
    }

    #pragma unroll
    for (int i = 0; i < 4; ++i) {
        const int crow0 = mbase + i * 16 + half * 8;
        #pragma unroll
        for (int j = 0; j < 2; ++j) {
            if (!nval[j]) continue;
            float* cptr = C + (size_t)crow0 * N + nbase + j * 16 + sub;
            #pragma unroll
            for (int r = 0; r < 8; ++r) cptr[(size_t)r * N] = acc[i][j][r];
        }
    }
}

// ---------------------------------------------------------------------------
// Elementwise helpers
// ---------------------------------------------------------------------------
__device__ __forceinline__ float silu_f(float x) {
    return x / (1.0f + __expf(-x));
}
__device__ __forceinline__ float softplus_f(float x) {
    return (x > 20.0f) ? x : log1pf(__expf(x));
}

__global__ void f32_to_bf16_kern(const float* __restrict__ src,
                                 bf16_t* __restrict__ dst, int n) {
    int i = blockIdx.x * blockDim.x + threadIdx.x;
    if (i < n) dst[i] = (bf16_t)src[i];
}

// Zero-padded convert: dst[r, 0..dstc) bf16 from src[r*sstride + soff + c] for c<valid.
__global__ void pad_convert_kern(const float* __restrict__ src, bf16_t* __restrict__ dst,
                                 int rows, int sstride, int soff, int valid, int dstc) {
    int i = blockIdx.x * blockDim.x + threadIdx.x;
    if (i >= rows * dstc) return;
    int r = i / dstc, c = i - r * dstc;
    float v = (c < valid) ? src[(size_t)r * sstride + soff + c] : 0.0f;
    dst[i] = (bf16_t)v;
}

// Causal depthwise conv (K=4) + bias + SiLU. x lives in xz columns [0, D_INNER).
__global__ void conv_silu_kern(const float* __restrict__ xz,
                               const float* __restrict__ w,
                               const float* __restrict__ bias,
                               float* __restrict__ xc, bf16_t* __restrict__ xcbf) {
    int i = blockIdx.x * blockDim.x + threadIdx.x;
    if (i >= MTOT * D_INNER) return;
    int d = i % D_INNER;
    int m = i / D_INNER;
    int t = m % SEQ, b = m / SEQ;
    float acc = bias[d];
    #pragma unroll
    for (int j = 0; j < D_CONV; ++j) {
        int tt = t - (D_CONV - 1) + j;
        if (tt >= 0)
            acc += w[d * D_CONV + j] * xz[((size_t)(b * SEQ + tt)) * (2 * D_INNER) + d];
    }
    float s = silu_f(acc);
    xc[i] = s;
    xcbf[i] = (bf16_t)s;
}

// ---------------------------------------------------------------------------
// Chunked selective scan
// Pass A: per (b, chunk, d): local scan from h=0 -> hend[34], sum of dt.
// ---------------------------------------------------------------------------
__global__ __launch_bounds__(256)
void scan_passA(const float* __restrict__ dt, const float* __restrict__ dt_bias,
                const float* __restrict__ xc, const float* __restrict__ xdbl,
                const float* __restrict__ A_log,
                float* __restrict__ hend, float* __restrict__ dtsum) {
    __shared__ float Bs[CHUNK][D_STATE];
    const int d = blockIdx.x * 256 + threadIdx.x;
    const int c = blockIdx.y, b = blockIdx.z;

    for (int idx = threadIdx.x; idx < CHUNK * D_STATE; idx += 256) {
        int tt = idx / D_STATE, n = idx % D_STATE;
        int m = b * SEQ + c * CHUNK + tt;
        Bs[tt][n] = xdbl[(size_t)m * XDCOLS + DT_RANK + n];
    }
    __syncthreads();

    float a[D_STATE], h[D_STATE];
    #pragma unroll
    for (int n = 0; n < D_STATE; ++n) {
        a[n] = -__expf(A_log[(size_t)d * D_STATE + n]);
        h[n] = 0.0f;
    }
    const float bias = dt_bias[d];
    float dsum = 0.0f;
    for (int tt = 0; tt < CHUNK; ++tt) {
        int m = b * SEQ + c * CHUNK + tt;
        float dtv = softplus_f(dt[(size_t)m * D_INNER + d] + bias);
        float u = xc[(size_t)m * D_INNER + d];
        float du = dtv * u;
        dsum += dtv;
        #pragma unroll
        for (int n = 0; n < D_STATE; ++n)
            h[n] = __expf(dtv * a[n]) * h[n] + du * Bs[tt][n];
    }
    size_t base = ((size_t)(b * NCHUNK + c) * D_INNER + d) * D_STATE;
    #pragma unroll
    for (int n = 0; n < D_STATE; ++n) hend[base + n] = h[n];
    dtsum[(size_t)(b * NCHUNK + c) * D_INNER + d] = dsum;
}

// Combine: sequential over the 32 chunks (cheap), parallel over (b, d).
// Hprev[b,c,d,:] = state entering chunk c; H_{c+1} = exp(a*sumdt_c)*H_c + hend_c.
__global__ void scan_combine(const float* __restrict__ A_log,
                             const float* __restrict__ dtsum,
                             const float* __restrict__ hend,
                             float* __restrict__ Hprev) {
    int id = blockIdx.x * blockDim.x + threadIdx.x;
    if (id >= BATCH * D_INNER) return;
    int b = id / D_INNER, d = id % D_INNER;
    float a[D_STATE], H[D_STATE];
    #pragma unroll
    for (int n = 0; n < D_STATE; ++n) {
        a[n] = -__expf(A_log[(size_t)d * D_STATE + n]);
        H[n] = 0.0f;
    }
    for (int c = 0; c < NCHUNK; ++c) {
        size_t base = ((size_t)(b * NCHUNK + c) * D_INNER + d) * D_STATE;
        float ds = dtsum[(size_t)(b * NCHUNK + c) * D_INNER + d];
        #pragma unroll
        for (int n = 0; n < D_STATE; ++n) {
            Hprev[base + n] = H[n];
            H[n] = __expf(a[n] * ds) * H[n] + hend[base + n];
        }
    }
}

// Pass B: replay chunk with correct initial state, produce y, fuse epilogue:
// y = (y + u*D_skip) * silu(z); store bf16 for the output WMMA GEMM.
__global__ __launch_bounds__(256)
void scan_passB(const float* __restrict__ dt, const float* __restrict__ dt_bias,
                const float* __restrict__ xc, const float* __restrict__ xdbl,
                const float* __restrict__ A_log, const float* __restrict__ Hprev,
                const float* __restrict__ D_skip, const float* __restrict__ xz,
                bf16_t* __restrict__ ybf) {
    __shared__ float Bs[CHUNK][D_STATE];
    __shared__ float Cs[CHUNK][D_STATE];
    const int d = blockIdx.x * 256 + threadIdx.x;
    const int c = blockIdx.y, b = blockIdx.z;

    for (int idx = threadIdx.x; idx < CHUNK * D_STATE; idx += 256) {
        int tt = idx / D_STATE, n = idx % D_STATE;
        int m = b * SEQ + c * CHUNK + tt;
        Bs[tt][n] = xdbl[(size_t)m * XDCOLS + DT_RANK + n];
        Cs[tt][n] = xdbl[(size_t)m * XDCOLS + DT_RANK + D_STATE + n];
    }
    __syncthreads();

    float a[D_STATE], h[D_STATE];
    size_t hbase = ((size_t)(b * NCHUNK + c) * D_INNER + d) * D_STATE;
    #pragma unroll
    for (int n = 0; n < D_STATE; ++n) {
        a[n] = -__expf(A_log[(size_t)d * D_STATE + n]);
        h[n] = Hprev[hbase + n];
    }
    const float bias = dt_bias[d];
    const float dsk = D_skip[d];
    for (int tt = 0; tt < CHUNK; ++tt) {
        int m = b * SEQ + c * CHUNK + tt;
        float dtv = softplus_f(dt[(size_t)m * D_INNER + d] + bias);
        float u = xc[(size_t)m * D_INNER + d];
        float du = dtv * u;
        float y = 0.0f;
        #pragma unroll
        for (int n = 0; n < D_STATE; ++n) {
            h[n] = __expf(dtv * a[n]) * h[n] + du * Bs[tt][n];
            y += h[n] * Cs[tt][n];
        }
        float z = xz[(size_t)m * (2 * D_INNER) + D_INNER + d];
        float out = (y + u * dsk) * silu_f(z);
        ybf[(size_t)m * D_INNER + d] = (bf16_t)out;
    }
}

// ---------------------------------------------------------------------------
// Host-side launch
// ---------------------------------------------------------------------------
extern "C" void kernel_launch(void* const* d_in, const int* in_sizes, int n_in,
                              void* d_out, int out_size, void* d_ws, size_t ws_size,
                              hipStream_t stream) {
    const float* hidden  = (const float*)d_in[0];
    const float* W_in    = (const float*)d_in[1];
    const float* conv_w  = (const float*)d_in[2];
    const float* conv_b  = (const float*)d_in[3];
    const float* W_x     = (const float*)d_in[4];
    const float* W_dt    = (const float*)d_in[5];
    const float* dt_bias = (const float*)d_in[6];
    const float* A_log   = (const float*)d_in[7];
    const float* D_skip  = (const float*)d_in[8];
    const float* W_out   = (const float*)d_in[9];
    float* out = (float*)d_out;

    char* w = (char*)d_ws;
    size_t off = 0;
    auto take = [&](size_t bytes) { void* p = w + off; off += (bytes + 255) & ~(size_t)255; return p; };

    bf16_t* hbf   = (bf16_t*)take((size_t)MTOT * D_MODEL * 2);
    bf16_t* wibf  = (bf16_t*)take((size_t)2 * D_INNER * D_MODEL * 2);
    float*  xz    = (float*) take((size_t)MTOT * 2 * D_INNER * 4);
    float*  xc    = (float*) take((size_t)MTOT * D_INNER * 4);
    bf16_t* xcbf  = (bf16_t*)take((size_t)MTOT * D_INNER * 2);
    bf16_t* wxbf  = (bf16_t*)take((size_t)XDCOLS * D_INNER * 2);
    float*  xdbl  = (float*) take((size_t)MTOT * XDCOLS * 4);
    bf16_t* dtrbf = (bf16_t*)take((size_t)MTOT * KPAD * 2);
    bf16_t* wdtbf = (bf16_t*)take((size_t)D_INNER * KPAD * 2);
    float*  dtbuf = (float*) take((size_t)MTOT * D_INNER * 4);
    bf16_t* ybf   = (bf16_t*)take((size_t)MTOT * D_INNER * 2);
    bf16_t* wobf  = (bf16_t*)take((size_t)D_MODEL * D_INNER * 2);
    float*  hend  = (float*) take((size_t)BATCH * NCHUNK * D_INNER * D_STATE * 4);
    float*  dtsum = (float*) take((size_t)BATCH * NCHUNK * D_INNER * 4);
    float*  Hprev = (float*) take((size_t)BATCH * NCHUNK * D_INNER * D_STATE * 4);
    (void)ws_size;

    const int T = 256;
    auto cdiv = [](int a, int b) { return (a + b - 1) / b; };

    // 1) fp32 -> bf16 conversions
    f32_to_bf16_kern<<<cdiv(MTOT * D_MODEL, T), T, 0, stream>>>(hidden, hbf, MTOT * D_MODEL);
    f32_to_bf16_kern<<<cdiv(2 * D_INNER * D_MODEL, T), T, 0, stream>>>(W_in, wibf, 2 * D_INNER * D_MODEL);
    f32_to_bf16_kern<<<cdiv(XDCOLS * D_INNER, T), T, 0, stream>>>(W_x, wxbf, XDCOLS * D_INNER);
    f32_to_bf16_kern<<<cdiv(D_MODEL * D_INNER, T), T, 0, stream>>>(W_out, wobf, D_MODEL * D_INNER);

    // 2) xz = hidden @ W_in^T   (M=4096, N=2560, K=640); 128x128 block tiles
    gemm_bf16_wmma<<<dim3(cdiv(2 * D_INNER, 128), MTOT / 128), T, 0, stream>>>(
        hbf, wibf, xz, MTOT, 2 * D_INNER, D_MODEL);

    // 3) causal dwconv + SiLU
    conv_silu_kern<<<cdiv(MTOT * D_INNER, T), T, 0, stream>>>(xz, conv_w, conv_b, xc, xcbf);

    // 4) x_dbl = xc @ W_x^T     (N=108, guarded)
    gemm_bf16_wmma<<<dim3(cdiv(XDCOLS, 128), MTOT / 128), T, 0, stream>>>(
        xcbf, wxbf, xdbl, MTOT, XDCOLS, D_INNER);

    // 5) dt = dt_raw @ W_dt^T   (K padded 40 -> 64)
    pad_convert_kern<<<cdiv(MTOT * KPAD, T), T, 0, stream>>>(xdbl, dtrbf, MTOT, XDCOLS, 0, DT_RANK, KPAD);
    pad_convert_kern<<<cdiv(D_INNER * KPAD, T), T, 0, stream>>>(W_dt, wdtbf, D_INNER, DT_RANK, 0, DT_RANK, KPAD);
    gemm_bf16_wmma<<<dim3(cdiv(D_INNER, 128), MTOT / 128), T, 0, stream>>>(
        dtrbf, wdtbf, dtbuf, MTOT, D_INNER, KPAD);

    // 6) chunked selective scan
    scan_passA<<<dim3(D_INNER / 256, NCHUNK, BATCH), T, 0, stream>>>(
        dtbuf, dt_bias, xc, xdbl, A_log, hend, dtsum);
    scan_combine<<<cdiv(BATCH * D_INNER, T), T, 0, stream>>>(A_log, dtsum, hend, Hprev);
    scan_passB<<<dim3(D_INNER / 256, NCHUNK, BATCH), T, 0, stream>>>(
        dtbuf, dt_bias, xc, xdbl, A_log, Hprev, D_skip, xz, ybf);

    // 7) out = y @ W_out^T      (M=4096, N=640, K=1280)
    gemm_bf16_wmma<<<dim3(cdiv(D_MODEL, 128), MTOT / 128), T, 0, stream>>>(
        ybf, wobf, out, MTOT, D_MODEL, D_INNER);
}
